// GATLayer_67963562492267
// MI455X (gfx1250) — matrix operand order repaired and checked by reading, not compiled, not run on hardware
//
#include <hip/hip_runtime.h>

typedef __attribute__((ext_vector_type(16))) __bf16 v16bf;
typedef __attribute__((ext_vector_type(8)))  float  v8f;
typedef __attribute__((ext_vector_type(2)))  float  v2f;

#define NN      4096
#define INDIM   128
#define OUTDIM  128
#define NHEAD   4
#define HDIM    32
#define NEGS    0.2f

// ---------------------------------------------------------------------------
// bf16 round-to-nearest-even helper
// ---------------------------------------------------------------------------
static __device__ __forceinline__ unsigned short f32_to_bf16(float f) {
    unsigned u = __float_as_uint(f);
    unsigned r = u + 0x7FFFu + ((u >> 16) & 1u);
    return (unsigned short)(r >> 16);
}

// ---------------------------------------------------------------------------
// Kernel 1: h = x @ W^T  via V_WMMA_F32_16X16X4_F32
// grid = 256 blocks (16 rows each), block = 256 thr = 8 waves (16 cols each)
// ---------------------------------------------------------------------------
__global__ __launch_bounds__(256) void h_wmma_kernel(
    const float* __restrict__ x, const float* __restrict__ W,
    float* __restrict__ hfeat) {
    __shared__ float xT[16][INDIM];
    const int i0 = blockIdx.x * 16;
    const int t  = threadIdx.x;
    {   // stage x tile: 16x128 f32, 8 floats per thread
        int f0 = t * 8;
        int row = f0 >> 7, col = f0 & 127;
        const float4* s = reinterpret_cast<const float4*>(x + (size_t)(i0 + row) * INDIM + col);
        *reinterpret_cast<float4*>(&xT[row][col])     = s[0];
        *reinterpret_cast<float4*>(&xT[row][col + 4]) = s[1];
    }
    __syncthreads();
    const int w = t >> 5, lane = t & 31;
    const int M = lane & 15, hi = lane >> 4;
    const int o0 = w * 16, Nn = o0 + M;
    v8f c = {};
    for (int k0 = 0; k0 < INDIM; k0 += 4) {
        int kk = k0 + 2 * hi;           // A elem v: K = v + 2*hi   (ISA 16x4 f32 layout)
        v2f a, b;
        a.x = xT[M][kk];  a.y = xT[M][kk + 1];
        b.x = W[(size_t)Nn * INDIM + kk];       // B elem v: K = v + 2*hi, N = lane&15
        b.y = W[(size_t)Nn * INDIM + kk + 1];
        c = __builtin_amdgcn_wmma_f32_16x16x4_f32(false, a, false, b, (short)0, c, false, false);
    }
#pragma unroll
    for (int r = 0; r < 8; ++r)   // C layout: row = r + 8*hi, col = lane&15
        hfeat[(size_t)(i0 + r + 8 * hi) * OUTDIM + o0 + M] = c[r];
}

// ---------------------------------------------------------------------------
// Kernel 2: e_src / e_dst + per-head encoded-float atomic max of e_dst
// ---------------------------------------------------------------------------
__global__ void e_kernel(const float* __restrict__ hfeat,
                         const float* __restrict__ asrc,
                         const float* __restrict__ adst,
                         float* __restrict__ e_src, float* __restrict__ e_dst,
                         unsigned* __restrict__ edmax) {
    int t  = blockIdx.x * blockDim.x + threadIdx.x;   // t = i*4 + hh
    int i  = t >> 2, hh = t & 3;
    const float* hp = hfeat + (size_t)i * OUTDIM + hh * HDIM;
    float s1 = 0.f, s2 = 0.f;
#pragma unroll
    for (int d = 0; d < HDIM; ++d) {
        float hv = hp[d];
        s1 += hv * asrc[hh * HDIM + d];
        s2 += hv * adst[hh * HDIM + d];
    }
    e_src[t] = s1;
    e_dst[t] = s2;
    unsigned u   = __float_as_uint(s2);                 // monotone unsigned encoding
    unsigned enc = u ^ ((u & 0x80000000u) ? 0xFFFFFFFFu : 0x80000000u);
    atomicMax(&edmax[hh], enc);
}

// ---------------------------------------------------------------------------
// Kernel 3: pack h (bf16) into the WMMA B-operand register layout:
//   packedB[hh][jt(128)][dt(2)][lane(32)][v(8)] : dword = {K=2v | K=2v+1}
//   with K = 16*(v>>2) + 8*(lane>>4) + 2*(v&3) (+1), N = lane&15  (ISA 32x16 bf16 B layout)
// ---------------------------------------------------------------------------
__global__ void packB_kernel(const float* __restrict__ hfeat,
                             unsigned* __restrict__ packedB) {
    int idx  = blockIdx.x * blockDim.x + threadIdx.x;   // 2^18 entries
    int v    = idx & 7;
    int lane = (idx >> 3) & 31;
    int dt   = (idx >> 8) & 1;
    int jt   = (idx >> 9) & 127;
    int hh   = idx >> 16;
    int hi = lane >> 4, Nn = lane & 15;
    int j = jt * 32 + 16 * (v >> 2) + 8 * hi + 2 * (v & 3);
    int d = hh * HDIM + dt * 16 + Nn;
    unsigned short lo   = f32_to_bf16(hfeat[(size_t)j * OUTDIM + d]);
    unsigned short hi16 = f32_to_bf16(hfeat[(size_t)(j + 1) * OUTDIM + d]);
    packedB[idx] = (unsigned)lo | ((unsigned)hi16 << 16);
}

// ---------------------------------------------------------------------------
// Kernel 4: fused masked-softmax attention + aggregation, single pass with
// precomputed stabilizer. 256 blocks x 8 waves (4 heads x 2 j-splits).
// adj / e_dst tiles are double-buffered in LDS and fetched with the CDNA5
// async-to-LDS path (ASYNCcnt), overlapping the next tile's HBM/L2 traffic
// with the exp + v_wmma_f32_16x16x32_bf16 work on the current tile.
// ---------------------------------------------------------------------------
__global__ __launch_bounds__(256) void attn_kernel(
    const int* __restrict__ adj,
    const float* __restrict__ e_src, const float* __restrict__ e_dst,
    const unsigned* __restrict__ edmax, const unsigned* __restrict__ packedB,
    float* __restrict__ out) {
    __shared__ int   adjT[2][2][16][32];    // [stage][j-half][row][col]
    __shared__ float edstT[2][2][32][4];    // [stage][j-half][j][head]
    __shared__ float lsumS[4][16];
    __shared__ float accbuf[4][16][32];

    const int i0 = blockIdx.x * 16;
    const int t  = threadIdx.x;
    const int w  = t >> 5, lane = t & 31;
    const int hh = w & 3,  jh   = w >> 2;
    const int M  = lane & 15, hi = lane >> 4;

    // per-row softmax stabilizer: m = lrelu(e_src[i] + max_j e_dst[j])
    float esrc_l = e_src[(size_t)(i0 + M) * NHEAD + hh];
    unsigned u  = edmax[hh];
    unsigned fu = (u & 0x80000000u) ? (u ^ 0x80000000u) : ~u;
    float edm = __uint_as_float(fu);
    float zb  = esrc_l + edm;
    float m_l = zb > 0.f ? zb : NEGS * zb;

    if (t < 64) lsumS[t >> 4][t & 15] = 0.f;

    // this thread's fixed staging assignment
    const int sS   = t >> 7;               // which j-half tile this thread loads
    const int eS   = t & 127;
    const int rowS = eS >> 3;
    const int c4S  = (eS & 7) << 2;
    const int jcS  = (t >> 2) & 31;
    const int hbS  = t & 3;

    // issue the async copies for tile-iteration `it2` into LDS stage it2&1
    auto issue_stage = [&](int it2) {
        int st  = it2 & 1;
        int j0s = (sS * 64 + it2) << 5;    // tile base j for this thread's half
        unsigned ldsA  = (unsigned)(uintptr_t)&adjT[st][sS][rowS][c4S];
        unsigned goffA = (unsigned)(((size_t)(i0 + rowS) * NN + j0s + c4S) * sizeof(int));
        asm volatile("global_load_async_to_lds_b128 %0, %1, %2"
                     :: "v"(ldsA), "v"(goffA), "s"(adj) : "memory");
        unsigned ldsE  = (unsigned)(uintptr_t)&edstT[st][sS][jcS][hbS];
        unsigned goffE = (unsigned)(((j0s + jcS) * NHEAD + hbS) * sizeof(float));
        asm volatile("global_load_async_to_lds_b32 %0, %1, %2"
                     :: "v"(ldsE), "v"(goffE), "s"(e_dst) : "memory");
    };

    v8f acc0 = {}, acc1 = {};
    float rowsum = 0.f;

    issue_stage(0);                        // prime the pipeline (2 async ops)
    for (int it = 0; it < 64; ++it) {
        __syncthreads();                   // buffer (it+1)&1 free for refill
        if (it + 1 < 64) {
            issue_stage(it + 1);           // prefetch next tile (ASYNCcnt -> 4)
            asm volatile("s_wait_asynccnt 0x2" ::: "memory");  // stage `it` landed
        } else {
            asm volatile("s_wait_asynccnt 0x0" ::: "memory");
        }
        __syncthreads();                   // all waves' portions of stage `it` in LDS
        const int st = it & 1;

        // Build A (16x32 bf16 layout: elem 8g+c <-> K = 16g + 8*hi + c)
        v16bf av;
        float ps = 0.f;
#pragma unroll
        for (int g = 0; g < 2; ++g) {
#pragma unroll
            for (int c = 0; c < 8; ++c) {
                int k   = (g << 4) + (hi << 3) + c;
                float z = esrc_l + edstT[st][jh][k][hh];
                float lr = z > 0.f ? z : NEGS * z;
                float p  = (adjT[st][jh][M][k] != 0) ? expf(lr - m_l) : 0.f;
                ps += p;
                av[(g << 3) + c] = (__bf16)p;
            }
        }
        rowsum += ps;

        int jt = (jh << 6) + it;
        const unsigned* bp = packedB + ((((size_t)hh * 128 + jt) * 2) * 32 + lane) * 8;
        v16bf b0 = *reinterpret_cast<const v16bf*>(bp);
        v16bf b1 = *reinterpret_cast<const v16bf*>(bp + 256);

        acc0 = __builtin_amdgcn_wmma_f32_16x16x32_bf16(false, av, false, b0, (short)0, acc0, false, false);
        acc1 = __builtin_amdgcn_wmma_f32_16x16x32_bf16(false, av, false, b1, (short)0, acc1, false, false);
    }

    // combine row sums (lane l and l+16 share a row; then both j-half waves)
    float rs = rowsum + __shfl_xor(rowsum, 16, 32);
    if (hi == 0) atomicAdd(&lsumS[hh][M], rs);
    if (jh == 1) {
#pragma unroll
        for (int r = 0; r < 8; ++r) {
            int Mr = r + (hi << 3);
            accbuf[hh][Mr][M]      = acc0[r];
            accbuf[hh][Mr][16 + M] = acc1[r];
        }
    }
    __syncthreads();
    if (jh == 0) {
#pragma unroll
        for (int r = 0; r < 8; ++r) {
            int Mr = r + (hi << 3);
            float tot = lsumS[hh][Mr];
            out[(size_t)(i0 + Mr) * OUTDIM + (hh << 5) + M]      = (acc0[r] + accbuf[hh][Mr][M]) / tot;
            out[(size_t)(i0 + Mr) * OUTDIM + (hh << 5) + 16 + M] = (acc1[r] + accbuf[hh][Mr][16 + M]) / tot;
        }
    }
}

// ---------------------------------------------------------------------------
extern "C" void kernel_launch(void* const* d_in, const int* in_sizes, int n_in,
                              void* d_out, int out_size, void* d_ws, size_t ws_size,
                              hipStream_t stream) {
    const float* x    = (const float*)d_in[0];
    const int*   adj  = (const int*)d_in[1];
    const float* W    = (const float*)d_in[2];
    const float* asrc = (const float*)d_in[3];
    const float* adst = (const float*)d_in[4];
    float* out = (float*)d_out;

    char* ws = (char*)d_ws;
    float*    hfeat   = (float*)(ws);                       // 4096*128*4  = 2 MB
    float*    e_src   = (float*)(ws + 2097152);             // 64 KB
    float*    e_dst   = (float*)(ws + 2162688);             // 64 KB
    unsigned* edmax   = (unsigned*)(ws + 2228224);          // 16 B
    unsigned* packedB = (unsigned*)(ws + 2228480);          // 1 MB

    hipMemsetAsync(edmax, 0, NHEAD * sizeof(unsigned), stream);   // encoded -inf
    h_wmma_kernel<<<NN / 16, 256, 0, stream>>>(x, W, hfeat);
    e_kernel<<<(NN * NHEAD) / 256, 256, 0, stream>>>(hfeat, asrc, adst, e_src, e_dst, edmax);
    packB_kernel<<<(NHEAD * 128 * 2 * 32 * 8) / 256, 256, 0, stream>>>(hfeat, packedB);
    attn_kernel<<<NN / 16, 256, 0, stream>>>(adj, e_src, e_dst, edmax, packedB, out);
}
